// FrameWiseLoss_26800595927495
// MI455X (gfx1250) — matrix operand order repaired and checked by reading, not compiled, not run on hardware
//
#include <hip/hip_runtime.h>
#include <hip/hip_bf16.h>

typedef __attribute__((ext_vector_type(16))) _Float16 v16h;
typedef __attribute__((ext_vector_type(8)))  float    v8f;

// Problem constants (from reference setup_inputs)
constexpr int S = 4, B = 8, C = 64, T = 16384;
constexpr int TCH        = 256;                 // t-chunk staged in LDS per iteration
constexpr int WGS_PER_SB = 16;                  // workgroups per (s,b)
constexpr int CHUNKS_PER_WG = T / TCH / WGS_PER_SB;  // 4

// ---------------------------------------------------------------------------
// Zero the workspace accumulators (sums[S*C*C], counts[C], nll[S]).
// ---------------------------------------------------------------------------
__global__ void fw_zero(float* __restrict__ ws, int n) {
  int i = blockIdx.x * blockDim.x + threadIdx.x;
  if (i < n) ws[i] = 0.0f;
}

// ---------------------------------------------------------------------------
// Main fused kernel:
//  - streams predictions once (coalesced along T, lane <-> t)
//  - per-(t) log-sum-exp over C=64 in registers (v_exp_f32 / v_log_f32)
//  - framewise NLL accumulated per lane -> LDS -> global atomic
//  - logp staged to LDS as f16; segment-sum by class done as
//    onehot(64xK) x logp(Kx64) with v_wmma_f32_16x16x32_f16,
//    f32 accumulators resident in VGPRs across all chunks.
// ---------------------------------------------------------------------------
__global__ __launch_bounds__(256) void fw_main(const float* __restrict__ pred,
                                               const int*   __restrict__ tgt,
                                               float* __restrict__ sums,
                                               float* __restrict__ counts,
                                               float* __restrict__ nll) {
  __shared__ _Float16 s_logp[C][TCH + 8];   // +8 halves padding vs bank conflicts
  __shared__ int      s_t[TCH];
  __shared__ _Float16 s_w[TCH];
  __shared__ float    s_cnt[C];
  __shared__ float    s_nll;

  const int tid  = threadIdx.x;
  const int lane = tid & 31;
  const int wave = tid >> 5;                 // 0..7

  const int wg    = blockIdx.x;
  const int s_idx = wg / (B * WGS_PER_SB);
  const int rem   = wg % (B * WGS_PER_SB);
  const int b     = rem / WGS_PER_SB;
  const int slot  = rem % WGS_PER_SB;

  if (tid < C) s_cnt[tid] = 0.0f;
  if (tid == 0) s_nll = 0.0f;

  // Each wave owns two 16x16 output tiles of the 64x64 class matrix.
  v8f acc0 = {}, acc1 = {};
  const int tile0 = wave * 2, tile1 = wave * 2 + 1;
  const int m0 = (tile0 >> 2) * 16, n0 = (tile0 & 3) * 16;
  const int m1 = (tile1 >> 2) * 16, n1 = (tile1 & 3) * 16;

  const float* predsb = pred + (size_t)(s_idx * B + b) * C * T;
  const int*   tgtb   = tgt + b * T;

  float nll_acc = 0.0f;

  for (int ch = 0; ch < CHUNKS_PER_WG; ++ch) {
    const int t0 = (slot + ch * WGS_PER_SB) * TCH;
    __syncthreads();  // LDS tile from previous iteration fully consumed

    // ---------------- Phase A: LSE + NLL + stage logp(f16) to LDS ----------
    const int tl = tid;            // local t in [0,256): 8 waves x 32 lanes
    const int t  = t0 + tl;
    const int tg = tgtb[t];
    const float w  = (tg >= 0) ? 1.0f : 0.0f;
    const int  tgs = (tg >= 0) ? tg : 0;
    s_t[tl] = tgs;
    s_w[tl] = (_Float16)w;

    float x[C];
    float m = -3.402823466e38f;
#pragma unroll
    for (int c = 0; c < C; ++c) {
      x[c] = predsb[(size_t)c * T + t];
      if ((c & 3) == 0)  // prefetch next chunk's cacheline for this class row
        __builtin_prefetch(&predsb[(size_t)c * T + t + WGS_PER_SB * TCH], 0, 1);
      m = fmaxf(m, x[c]);
    }
    float sum = 0.0f;
#pragma unroll
    for (int c = 0; c < C; ++c) sum += __expf(x[c] - m);
    const float lse = m + __logf(sum);
    // x[tgs] re-load: the line is guaranteed hot in cache; this avoids a
    // 64-way v_cndmask select chain for the dynamic register index.
    const float xt = predsb[(size_t)tgs * T + t];
    nll_acc += w * (lse - xt);
#pragma unroll
    for (int c = 0; c < C; ++c) s_logp[c][tl] = (_Float16)(x[c] - lse);
    if (s_idx == 0) atomicAdd(&s_cnt[tgs], w);   // ds_add_f32
    __syncthreads();

    // ---------------- Phase B: onehot x logp via WMMA ----------------------
    const int hi = lane >> 4;
    const int ln = lane & 15;
#pragma unroll 1
    for (int ks = 0; ks < TCH / 32; ++ks) {
      const int k0 = ks * 32;
      v16h a0, a1, bv0, bv1;
      // A (16x32 f16, one row per lane): one-hot of target, weighted by w.
#pragma unroll
      for (int j = 0; j < 16; ++j) {
        const int K = k0 + j + 8 * hi + ((j >= 8) ? 8 : 0);
        const _Float16 wv = s_w[K];
        const int tv = s_t[K];
        a0[j] = (tv == m0 + ln) ? wv : (_Float16)0.0f;
        a1[j] = (tv == m1 + ln) ? wv : (_Float16)0.0f;
      }
      // B (32x16 f16): column = class ln within tile, K along t.
#pragma unroll
      for (int j = 0; j < 16; ++j) {
        const int K = k0 + j + 8 * hi + ((j >= 8) ? 8 : 0);
        bv0[j] = s_logp[n0 + ln][K];
        bv1[j] = s_logp[n1 + ln][K];
      }
      acc0 = __builtin_amdgcn_wmma_f32_16x16x32_f16(false, a0, false, bv0,
                                                    (short)0, acc0, false, false);
      acc1 = __builtin_amdgcn_wmma_f32_16x16x32_f16(false, a1, false, bv1,
                                                    (short)0, acc1, false, false);
    }
  }

  // ---------------- Flush accumulators -------------------------------------
  {
    const int hi = lane >> 4, ln = lane & 15;
    float* base = sums + (size_t)s_idx * C * C;
#pragma unroll
    for (int r = 0; r < 8; ++r) {
      atomicAdd(&base[(m0 + r + 8 * hi) * C + (n0 + ln)], acc0[r]);
      atomicAdd(&base[(m1 + r + 8 * hi) * C + (n1 + ln)], acc1[r]);
    }
  }
  atomicAdd(&s_nll, nll_acc);
  __syncthreads();
  if (tid == 0) atomicAdd(&nll[s_idx], s_nll);
  if (s_idx == 0 && tid < C) atomicAdd(&counts[tid], s_cnt[tid]);
}

// ---------------------------------------------------------------------------
// Finalize: tiny epilogue, single thread (4*64 reads).
// ---------------------------------------------------------------------------
__global__ void fw_final(const float* __restrict__ sums,
                         const float* __restrict__ counts,
                         const float* __restrict__ nll,
                         float* __restrict__ out) {
  if (threadIdx.x != 0 || blockIdx.x != 0) return;
  float n_valid = 0.0f, n_present = 0.0f;
  for (int c = 0; c < C; ++c) {
    n_valid += counts[c];
    n_present += (counts[c] > 0.0f) ? 1.0f : 0.0f;
  }
  if (n_present < 1.0f) n_present = 1.0f;
  float fw = 0.0f, fwg = 0.0f;
  for (int s = 0; s < S; ++s) {
    fw += nll[s] / n_valid;
    float dsum = 0.0f;
    for (int c = 0; c < C; ++c) {
      const float cc = counts[c];
      const float pres = (cc > 0.0f) ? 1.0f : 0.0f;
      dsum += pres * sums[(s * C + c) * C + c] / fmaxf(cc, 1.0f);
    }
    fwg += -dsum / n_present;
  }
  out[0] = fw + fwg / (float)S;
}

// ---------------------------------------------------------------------------
extern "C" void kernel_launch(void* const* d_in, const int* in_sizes, int n_in,
                              void* d_out, int out_size, void* d_ws, size_t ws_size,
                              hipStream_t stream) {
  const float* pred = (const float*)d_in[0];   // [S,B,C,T] fp32
  const int*   tgt  = (const int*)d_in[1];     // [B,T] int32
  // d_in[2] = mask (unused by reference), d_in[3] = epoch (unused)
  float* out = (float*)d_out;

  float* ws     = (float*)d_ws;
  float* sums   = ws;                     // S*C*C
  float* counts = ws + S * C * C;         // C
  float* nllv   = counts + C;             // S
  const int nz = S * C * C + C + S;

  fw_zero<<<(nz + 255) / 256, 256, 0, stream>>>(ws, nz);
  fw_main<<<S * B * WGS_PER_SB, 256, 0, stream>>>(pred, tgt, sums, counts, nllv);
  fw_final<<<1, 1, 0, stream>>>(sums, counts, nllv, out);
}
